// Attention_73323681677441
// MI455X (gfx1250) — matrix operand order repaired and checked by reading, not compiled, not run on hardware
//
#include <hip/hip_runtime.h>

// ---------------------------------------------------------------------------
// Fused single-head attention for B=4, C=64, H=W=64 (N=4096) on gfx1250.
//  Pass 1: QKV 1x1-conv projections (f32 math, bf16 outputs, layouts for WMMA)
//  Pass 2: flash-attention, one wave per 16-token tile, bf16 WMMA, f32 accum.
//          j-step = 64 keys/iteration: 16 WMMAs per iteration, softmax fixed
//          costs (shuffles, O-rescale, bookkeeping) amortized 2x vs j-step 32.
// ---------------------------------------------------------------------------

#define NTOK 4096
#define NCH  64

typedef __attribute__((ext_vector_type(16))) __bf16        v16bf;
typedef __attribute__((ext_vector_type(8)))  float         v8f;
typedef __attribute__((ext_vector_type(4)))  unsigned int  u32x4;

union FragBF {
    v16bf        bf;     // 8 VGPRs = A or B fragment for 16x16x32 bf16 WMMA
    u32x4        q[2];
    unsigned int u[8];
};

__device__ __forceinline__ u32x4 ld128(const unsigned short* p) {
    return *(const u32x4*)p;
}

__device__ __forceinline__ unsigned int f32_to_bf16(float f) {
    unsigned int x = __float_as_uint(f);
    return (x + 0x7FFFu + ((x >> 16) & 1u)) >> 16;   // round-to-nearest-even
}
__device__ __forceinline__ unsigned int pack2bf(float lo, float hi) {
    return f32_to_bf16(lo) | (f32_to_bf16(hi) << 16);
}

__device__ __forceinline__ v8f wmma_bf16(const FragBF& A, const FragBF& B, v8f C) {
    return __builtin_amdgcn_wmma_f32_16x16x32_bf16(
        /*neg_a=*/false, A.bf, /*neg_b=*/false, B.bf,
        /*c_mod=*/(short)0, C, /*reuse_a=*/false, /*reuse_b=*/false);
}

// ---------------------------------------------------------------------------
// Kernel 1: q/k/v projections.  gridDim.z selects projection:
//   z=0 -> Q, written transposed  QT[b][n][c]   (B-fragment feed)
//   z=1 -> K, written transposed  KT[b][n][c]   (A-fragment feed)
//   z=2 -> V, written natural     V [b][c][n]   (A-fragment feed)
// ---------------------------------------------------------------------------
__global__ __launch_bounds__(256) void qkv_proj_kernel(
    const float* __restrict__ x,
    const float* __restrict__ wq, const float* __restrict__ bq,
    const float* __restrict__ wk, const float* __restrict__ bk,
    const float* __restrict__ wv, const float* __restrict__ bv,
    unsigned short* __restrict__ QT,
    unsigned short* __restrict__ KT,
    unsigned short* __restrict__ Vn)
{
    const int b = blockIdx.y;
    const int z = blockIdx.z;
    const float* w    = (z == 0) ? wq : (z == 1) ? wk : wv;
    const float* bias = (z == 0) ? bq : (z == 1) ? bk : bv;

    const int idx = blockIdx.x * 256 + threadIdx.x;   // over 64*4096 outputs
    const int o   = idx >> 12;                        // output channel
    const int n   = idx & (NTOK - 1);                 // token

    const float* xp = x + (size_t)b * NCH * NTOK + n;
    const float* wr = w + o * NCH;

    float acc = bias[o];
#pragma unroll
    for (int c = 0; c < NCH; ++c)
        acc = fmaf(wr[c], xp[(size_t)c * NTOK], acc);

    const unsigned short h = (unsigned short)f32_to_bf16(acc);
    if (z == 2) {
        Vn[((size_t)b * NCH + o) * NTOK + n] = h;
    } else {
        unsigned short* dst = (z == 0) ? QT : KT;
        dst[((size_t)b * NTOK + n) * NCH + o] = h;
    }
}

// ---------------------------------------------------------------------------
// Kernel 2: fused flash attention.  256 threads = 8 waves; each wave owns one
// 16-token i-tile.  Sᵀ = Kᵀ·Q so softmax runs along VGPRs (per lane), and the
// exp() tile is already in B-fragment orientation for the V·Pᵀ WMMA.
// ---------------------------------------------------------------------------
__global__ __launch_bounds__(256) void attn_flash_kernel(
    const unsigned short* __restrict__ QT,
    const unsigned short* __restrict__ KT,
    const unsigned short* __restrict__ Vn,
    float* __restrict__ out)
{
    const int lane = threadIdx.x & 31;
    const int wave = threadIdx.x >> 5;
    const int tile = blockIdx.x * 8 + wave;   // 0 .. B*256-1
    const int b    = tile >> 8;
    const int i0   = (tile & 255) << 4;       // token-tile base
    const int col  = lane & 15;               // i column owned by this lane
    const int half = lane >> 4;               // lane-half (K-split of frags)

    const unsigned short* kbase = KT + (size_t)b * NTOK * NCH;
    const unsigned short* vbase = Vn + (size_t)b * NCH * NTOK;

    // ---- Q B-fragments: B[kk=c][n=i]; lane holds column i, c packed in VGPRs
    FragBF Bq0, Bq1;
    {
        const unsigned short* qrow =
            QT + ((size_t)b * NTOK + i0 + col) * NCH + half * 8;
        Bq0.q[0] = ld128(qrow);        Bq0.q[1] = ld128(qrow + 16);  // c 0..31
        Bq1.q[0] = ld128(qrow + 32);   Bq1.q[1] = ld128(qrow + 48);  // c 32..63
    }

    v8f O[4];
#pragma unroll
    for (int ct = 0; ct < 4; ++ct) O[ct] = (v8f){0,0,0,0,0,0,0,0};
    float m_run = -3.0e30f, l_run = 0.0f;
    const float scale = 1.0f / (float)NTOK;   // attn logits scaled by 1/N

    for (int j0 = 0; j0 < NTOK; j0 += 64) {
        // ---- Kᵀ A-fragments: A[m=j][kk=c], 4 j-subtiles x 2 c-blocks
        FragBF A0[4], A1[4];
#pragma unroll
        for (int jt = 0; jt < 4; ++jt) {
            const unsigned short* p =
                kbase + (size_t)(j0 + jt * 16 + col) * NCH + half * 8;
            A0[jt].q[0] = ld128(p);       A0[jt].q[1] = ld128(p + 16);  // c 0..31
            A1[jt].q[0] = ld128(p + 32);  A1[jt].q[1] = ld128(p + 48);  // c 32..63
        }

        if (j0 + 64 < NTOK) {   // prefetch next K/V tiles -> global_prefetch_b8
            __builtin_prefetch(kbase + (size_t)(j0 + 64 + col) * NCH, 0, 3);
            __builtin_prefetch(vbase + (size_t)col * NTOK + j0 + 64, 0, 3);
        }

        // ---- Sᵀ tiles: D[m=j][n=i]  (j in VGPRs/lane-half, i in lanes)
        v8f S[4];
#pragma unroll
        for (int jt = 0; jt < 4; ++jt) {
            v8f acc = (v8f){0,0,0,0,0,0,0,0};
            acc = wmma_bf16(A0[jt], Bq0, acc);
            acc = wmma_bf16(A1[jt], Bq1, acc);
            S[jt] = acc;
        }

        // ---- online softmax over this 64-wide j slab (per column i)
        float s[32];
#pragma unroll
        for (int jt = 0; jt < 4; ++jt)
#pragma unroll
            for (int r = 0; r < 8; ++r)
                s[jt * 8 + r] = S[jt][r] * scale;

        float mt = s[0];
#pragma unroll
        for (int r = 1; r < 32; ++r) mt = fmaxf(mt, s[r]);
        mt = fmaxf(mt, __shfl_xor(mt, 16, 32));          // merge lane halves
        const float m_new = fmaxf(m_run, mt);
        const float alpha = __expf(m_run - m_new);

        float p[32], psum = 0.0f;
#pragma unroll
        for (int r = 0; r < 32; ++r) { p[r] = __expf(s[r] - m_new); psum += p[r]; }
        psum += __shfl_xor(psum, 16, 32);
        l_run = l_run * alpha + psum;
        m_run = m_new;

#pragma unroll
        for (int ct = 0; ct < 4; ++ct)
#pragma unroll
            for (int r = 0; r < 8; ++r) O[ct][r] *= alpha;

        // ---- build Pᵀ B-fragments (kk=j, n=i): only lane-half swaps needed
        unsigned int pk[4][4];
#pragma unroll
        for (int jt = 0; jt < 4; ++jt)
#pragma unroll
            for (int k = 0; k < 4; ++k)
                pk[jt][k] = pack2bf(p[jt * 8 + 2 * k], p[jt * 8 + 2 * k + 1]);

        FragBF Bp0, Bp1;                      // j 0..31 and j 32..63 of slab
#pragma unroll
        for (int k = 0; k < 4; ++k) {
            const unsigned int x0 = (unsigned int)__shfl_xor((int)pk[0][k], 16, 32);
            const unsigned int x1 = (unsigned int)__shfl_xor((int)pk[1][k], 16, 32);
            const unsigned int x2 = (unsigned int)__shfl_xor((int)pk[2][k], 16, 32);
            const unsigned int x3 = (unsigned int)__shfl_xor((int)pk[3][k], 16, 32);
            Bp0.u[k]     = half ? x1       : pk[0][k];   // K rows 0..7 (half 0), 16..23 (half 1)
            Bp0.u[k + 4] = half ? pk[1][k] : x0;         // K rows 8..15 (half 0), 24..31 (half 1)
            Bp1.u[k]     = half ? x3       : pk[2][k];
            Bp1.u[k + 4] = half ? pk[3][k] : x2;
        }

        // ---- O(c-tile) += V(c-tile, j-slab) x Pᵀ : 8 WMMAs
#pragma unroll
        for (int ct = 0; ct < 4; ++ct) {
            const unsigned short* vp =
                vbase + (size_t)(ct * 16 + col) * NTOK + j0 + half * 16;
            FragBF Av;
            Av.q[0] = ld128(vp);      Av.q[1] = ld128(vp + 8);    // j 0..31
            O[ct] = wmma_bf16(Av, Bp0, O[ct]);
            Av.q[0] = ld128(vp + 32); Av.q[1] = ld128(vp + 40);   // j 32..63
            O[ct] = wmma_bf16(Av, Bp1, O[ct]);
        }
    }

    // ---- normalize and store: out[b][c][i], c = ct*16 + half*8 + r, i = i0+col
    const float rl = 1.0f / l_run;
    float* op = out + ((size_t)b * NCH + half * 8) * NTOK + i0 + col;
#pragma unroll
    for (int ct = 0; ct < 4; ++ct)
#pragma unroll
        for (int r = 0; r < 8; ++r)
            op[(size_t)(ct * 16 + r) * NTOK] = O[ct][r] * rl;
}

// ---------------------------------------------------------------------------
extern "C" void kernel_launch(void* const* d_in, const int* in_sizes, int n_in,
                              void* d_out, int out_size, void* d_ws, size_t ws_size,
                              hipStream_t stream)
{
    const float* x  = (const float*)d_in[0];
    const float* wq = (const float*)d_in[1];
    const float* bq = (const float*)d_in[2];
    const float* wk = (const float*)d_in[3];
    const float* bk = (const float*)d_in[4];
    const float* wv = (const float*)d_in[5];
    const float* bv = (const float*)d_in[6];
    float* out = (float*)d_out;

    const size_t per_buf = (size_t)4 * NTOK * NCH;        // elements (bf16)
    unsigned short* QT = (unsigned short*)d_ws;           // QT[b][n][c]
    unsigned short* KT = QT + per_buf;                    // KT[b][n][c]
    unsigned short* Vn = KT + per_buf;                    // V [b][c][n]

    // Pass 1: projections (1024 blocks per (batch, proj))
    dim3 g1((NCH * NTOK) / 256, 4, 3);
    qkv_proj_kernel<<<g1, 256, 0, stream>>>(x, wq, bq, wk, bk, wv, bv, QT, KT, Vn);

    // Pass 2: fused flash attention. 1024 wave-tiles, 8 waves per block.
    attn_flash_kernel<<<128, 256, 0, stream>>>(QT, KT, Vn, out);
}
